// TowerGNN_86672440033855
// MI455X (gfx1250) — matrix operand order repaired and checked by reading, not compiled, not run on hardware
//
#include <hip/hip_runtime.h>

#define NND 32768   // nodes
#define NE  65536   // edges
#define NG  1024    // graphs
#define HID 200
#define NTW 8
#define TDM 25

typedef __attribute__((ext_vector_type(16))) __bf16 v16bf;
typedef __attribute__((ext_vector_type(8)))  __bf16 v8bf;
typedef __attribute__((ext_vector_type(8)))  float  v8f;

union abf16 { v16bf v; v8bf h[2]; };

__device__ __forceinline__ float silu_(float x) { return x / (1.0f + __expf(-x)); }
__device__ __forceinline__ float sigm_(float x) { return 1.0f / (1.0f + __expf(-x)); }
__device__ __forceinline__ float wave_sum(float v) {
  #pragma unroll
  for (int o = 16; o > 0; o >>= 1) v += __shfl_xor(v, o);
  return v;
}
__device__ __forceinline__ unsigned foenc(float f) {
  unsigned u = __float_as_uint(f);
  return (u & 0x80000000u) ? ~u : (u | 0x80000000u);
}
__device__ __forceinline__ float fodec(unsigned u) {
  unsigned v = (u & 0x80000000u) ? (u & 0x7FFFFFFFu) : ~u;
  return __uint_as_float(v);
}

// ---------------------------------------------------------------- utilities
__global__ void zero_f32_kernel(float* p, long n) {
  long id = (long)blockIdx.x * blockDim.x + threadIdx.x;
  if (id < n) p[id] = 0.0f;
}
// pad-convert f32 row-major (rows x Kin) -> bf16 (rows x Kpad), zero pad cols
__global__ void pad_bf16_kernel(const float* __restrict__ in, __bf16* __restrict__ out,
                                long rows, int Kin, int Kpad) {
  long id = (long)blockIdx.x * blockDim.x + threadIdx.x;
  if (id >= rows * (long)Kpad) return;
  int c = (int)(id % Kpad);
  long r = id / Kpad;
  out[id] = (__bf16)((c < Kin) ? in[r * Kin + c] : 0.0f);
}
// stage B (K x N row-major f32, batched) into WMMA fragment order:
// Bs[((z*nKt + kt)*nTiles + nt)*512 + lane*16 + j]
//   lane<16: col = nt*16+lane, k = kt*32 + j ; lane>=16: col = nt*16+lane-16, k = kt*32+16+j
__global__ void stage_b_kernel(const float* __restrict__ B, __bf16* __restrict__ Bs,
                               int K, int N, int nKt, int nTiles, long sBin, long total) {
  long id = (long)blockIdx.x * blockDim.x + threadIdx.x;
  if (id >= total) return;
  int j = (int)(id & 15);
  int lane = (int)((id >> 4) & 31);
  long rest = id >> 9;
  int nt = (int)(rest % nTiles);
  long rest2 = rest / nTiles;
  int kt = (int)(rest2 % nKt);
  int z = (int)(rest2 / nKt);
  int c = nt * 16 + (lane & 15);
  int k = kt * 32 + ((lane >> 4) << 4) + j;
  float v = 0.0f;
  if (k < K && c < N) v = B[(long)z * sBin + (long)k * N + c];
  Bs[id] = (__bf16)v;
}

// ---------------------------------------------------------------- degree
__global__ void deg_kernel(const int* __restrict__ eidx, float* __restrict__ deg) {
  int e = blockIdx.x * blockDim.x + threadIdx.x;
  if (e < NE) atomicAdd(&deg[eidx[NE + e]], 1.0f);
}

// ---------------------------------------------------------------- h0 = silu(x@fw+fb)
__global__ void h0_kernel(const float* __restrict__ x, const float* __restrict__ fw,
                          const float* __restrict__ fb, float* __restrict__ state) {
  long id = (long)blockIdx.x * blockDim.x + threadIdx.x;
  if (id >= (long)NND * HID) return;
  int n = (int)(id / HID), j = (int)(id % HID);
  float acc = fb[j];
  const float* xp = x + (size_t)n * 11;
  #pragma unroll
  for (int i = 0; i < 11; ++i) acc += xp[i] * fw[i * HID + j];
  state[id] = silu_(acc);
}

// ---------------------------------------------------------------- edge MLP linear1
__global__ void edge_lin1_kernel(const float* __restrict__ ea, const float* __restrict__ w1,
                                 const float* __restrict__ b1, float* __restrict__ e_lin) {
  long id = (long)blockIdx.x * blockDim.x + threadIdx.x;
  if (id >= (long)NTW * NE * 50) return;
  int c = (int)(id % 50);
  long te = id / 50;
  int e = (int)(te % NE);
  int t = (int)(te / NE);
  float acc = b1[t * 50 + c];
  const float* ep = ea + (size_t)e * 4;
  const float* wp = w1 + t * 200 + c;
  #pragma unroll
  for (int i = 0; i < 4; ++i) acc += ep[i] * wp[i * 50];
  e_lin[id] = acc;
}

// ---------------------------------------------------------------- batchnorm statistics
__global__ void bn_stats_kernel(const float* __restrict__ data, float* __restrict__ mean,
                                float* __restrict__ var, int count, int inner, long outer) {
  int ch = blockIdx.x;
  int c = ch % inner, g = ch / inner;
  const float* p = data + (long)g * outer + c;
  float s = 0.f, s2 = 0.f;
  for (int i = threadIdx.x; i < count; i += blockDim.x) {
    float v = p[(long)i * inner];
    s += v; s2 += v * v;
  }
  __shared__ float sh[256], sh2[256];
  sh[threadIdx.x] = s; sh2[threadIdx.x] = s2;
  __syncthreads();
  for (int st = 128; st > 0; st >>= 1) {
    if (threadIdx.x < st) { sh[threadIdx.x] += sh[threadIdx.x + st]; sh2[threadIdx.x] += sh2[threadIdx.x + st]; }
    __syncthreads();
  }
  if (threadIdx.x == 0) {
    float ic = 1.0f / (float)count;
    float m = sh[0] * ic;
    mean[ch] = m;
    var[ch] = sh2[0] * ic - m * m;
  }
}

// ---------------------------------------------------------------- edge BN + SiLU -> bf16 (padded to 64 cols)
__global__ void edge_bn_silu_kernel(const float* __restrict__ e_lin, const float* __restrict__ mean,
                                    const float* __restrict__ var, const float* __restrict__ g1,
                                    const float* __restrict__ be1, __bf16* __restrict__ e_pad) {
  long id = (long)blockIdx.x * blockDim.x + threadIdx.x;
  if (id >= (long)NTW * NE * 64) return;
  int c = (int)(id & 63);
  long te = id >> 6;
  int e = (int)(te % NE);
  int t = (int)(te / NE);
  float r = 0.0f;
  if (c < 50) {
    int ch = t * 50 + c;
    float v = e_lin[((long)t * NE + e) * 50 + c];
    r = silu_(g1[ch] * (v - mean[ch]) * rsqrtf(var[ch] + 1e-5f) + be1[ch]);
  }
  e_pad[id] = (__bf16)r;
}

// ---------------------------------------------------------------- mixing BN + SiLU -> bf16 (padded to 224 cols)
__global__ void bn_silu_bf16_kernel(const float* __restrict__ t1, const float* __restrict__ mean,
                                    const float* __restrict__ var, const float* __restrict__ g,
                                    const float* __restrict__ b, __bf16* __restrict__ outb) {
  long id = (long)blockIdx.x * blockDim.x + threadIdx.x;
  if (id >= (long)NND * 224) return;
  int j = (int)(id % 224);
  float r = 0.0f;
  if (j < HID) {
    long row = id / 224;
    float v = t1[row * HID + j];
    r = silu_(g[j] * (v - mean[j]) * rsqrtf(var[j] + 1e-5f) + b[j]);
  }
  outb[id] = (__bf16)r;
}

// ---------------------------------------------------------------- WMMA bf16 GEMM
// A: (M x NKT*32) bf16 row-major, K pre-padded -> branch-free, vectorized fragment loads.
// B: pre-swizzled fragment layout in global; the block's 4-tile x NKT panel is staged
//    once into LDS cooperatively (shared by all 8 waves), WMMA reads B via ds_load_b128.
__device__ __forceinline__ v16bf load_a_frag(const __bf16* __restrict__ A, int lda, int row,
                                             int kk, int lane) {
  // 16-bit A 16x32 (matches staged B): lanes0-15 M=lane,K{0..15}; lanes16-31 K{16..31}
  int r = row + (lane & 15);
  int kb = kk + ((lane >> 4) << 4);
  const __bf16* p = A + (long)r * lda + kb;
  abf16 u;
  u.h[0] = *(const v8bf*)(p);
  u.h[1] = *(const v8bf*)(p + 8);
  return u.v;
}

template <bool OUT_BF16, int NKT>
__global__ __launch_bounds__(256) void wmma_gemm_kernel(
    const __bf16* __restrict__ A, const __bf16* __restrict__ Bs,
    const float* __restrict__ bias, void* __restrict__ Cout,
    int M, int N, int nTiles, long sA, long sB, long sBias, long sC) {
  __shared__ __bf16 bsh[NKT * 4 * 512];  // block's B panel: NKT k-tiles x 4 n-tiles
  int z = blockIdx.z;
  const __bf16* Az = A + (long)z * sA;
  const __bf16* Bz = Bs + (long)z * sB;
  const float* bz = bias ? (bias + (long)z * sBias) : nullptr;
  int tid = threadIdx.x;
  int lane = tid & 31;
  int wave = tid >> 5;
  int row = (blockIdx.y * 8 + wave) * 16;
  int col0 = blockIdx.x * 64;
  int nt0 = blockIdx.x * 4;
  const int Kpad = NKT * 32;

  // cooperative stage of the B panel (16B units); layout stays fragment-linear
  #pragma unroll
  for (int u = tid; u < NKT * 256; u += 256) {
    int s = u >> 1;          // fragment slot: [kt][tl][lane]
    int kt = s >> 7;
    int rem = s & 127;
    int tl = rem >> 5;
    int ln = rem & 31;
    long gbase = ((((long)kt * nTiles + nt0 + tl) * 32 + ln) << 4) + (long)(u & 1) * 8;
    *(v8bf*)(bsh + (long)u * 8) = *(const v8bf*)(Bz + gbase);
  }
  __syncthreads();

  v8f acc[4];
  #pragma unroll
  for (int t = 0; t < 4; ++t)
    #pragma unroll
    for (int r = 0; r < 8; ++r) acc[t][r] = 0.0f;

  // A stream: double-buffered global fragments (B comes from LDS)
  v16bf a_cur = load_a_frag(Az, Kpad, row, 0, lane);
  #pragma unroll
  for (int kt = 0; kt < NKT; ++kt) {
    int ktn = (kt + 1 < NKT) ? (kt + 1) : kt;
    v16bf a_nxt = load_a_frag(Az, Kpad, row, ktn << 5, lane);
    const __bf16* bp = bsh + (((kt * 4) * 32 + lane) << 4);
    v16bf b0 = *(const v16bf*)(bp);
    v16bf b1 = *(const v16bf*)(bp + 512);
    v16bf b2 = *(const v16bf*)(bp + 1024);
    v16bf b3 = *(const v16bf*)(bp + 1536);
    acc[0] = __builtin_amdgcn_wmma_f32_16x16x32_bf16(false, a_cur, false, b0, (short)0, acc[0], false, false);
    acc[1] = __builtin_amdgcn_wmma_f32_16x16x32_bf16(false, a_cur, false, b1, (short)0, acc[1], false, false);
    acc[2] = __builtin_amdgcn_wmma_f32_16x16x32_bf16(false, a_cur, false, b2, (short)0, acc[2], false, false);
    acc[3] = __builtin_amdgcn_wmma_f32_16x16x32_bf16(false, a_cur, false, b3, (short)0, acc[3], false, false);
    a_cur = a_nxt;
  }
  // D layout: lanes0-15 N=lane, M=row+r; lanes16-31 N=lane-16, M=row+8+r
  int mr = row + ((lane >> 4) << 3);
  int nc0 = (lane & 15);
  #pragma unroll
  for (int t = 0; t < 4; ++t) {
    int n = col0 + t * 16 + nc0;
    bool nok = (n < N);
    float bv = (bz && nok) ? bz[n] : 0.0f;
    #pragma unroll
    for (int r = 0; r < 8; ++r) {
      int m = mr + r;
      if (nok && m < M) {
        float v = acc[t][r] + bv;
        long idx = (long)z * sC + (long)m * N + n;
        if (OUT_BF16) ((__bf16*)Cout)[idx] = (__bf16)v;
        else          ((float*)Cout)[idx] = v;
      }
    }
  }
}

// ---------------------------------------------------------------- message: msg_e = x_src @ We_e, aggregated at dst
__global__ void msg_kernel(const float* __restrict__ state, const __bf16* __restrict__ We,
                           const int* __restrict__ eidx, float* __restrict__ agg) {
  int gw = (int)(((long)blockIdx.x * blockDim.x + threadIdx.x) >> 5);
  int lane = threadIdx.x & 31;
  if (gw >= NE * NTW) return;
  int t = gw & 7;
  int e = gw >> 3;
  int src = eidx[e];
  int dst = eidx[NE + e];
  int d = (lane < TDM) ? lane : 0;
  const __bf16* w = We + ((size_t)t * NE + e) * 625 + d;
  const float* sp = state + (size_t)src * HID + t * TDM;
  float acc = 0.0f;
  #pragma unroll
  for (int i = 0; i < TDM; ++i) acc += sp[i] * (float)w[i * TDM];
  if (lane < TDM) atomicAdd(&agg[(size_t)dst * HID + t * TDM + lane], acc);
}

// ---------------------------------------------------------------- NNConv-root + GRU, wave per node
__global__ void node_update_kernel(float* __restrict__ state, const float* __restrict__ agg,
                                   const float* __restrict__ deg, const float* __restrict__ root,
                                   const float* __restrict__ cb, const float* __restrict__ Wih,
                                   const float* __restrict__ bih, const float* __restrict__ Whh,
                                   const float* __restrict__ bhh) {
  int n = (int)(((long)blockIdx.x * blockDim.x + threadIdx.x) >> 5);
  int lane = threadIdx.x & 31;
  if (n >= NND) return;
  float dinv = 1.0f / fmaxf(deg[n], 1.0f);
  int d = (lane < TDM) ? lane : 0;
  for (int t = 0; t < NTW; ++t) {
    float h = state[(size_t)n * HID + t * TDM + d];
    float acc = agg[(size_t)n * HID + t * TDM + d] * dinv + cb[t * TDM + d];
    const float* rt = root + t * 625;
    #pragma unroll
    for (int i = 0; i < TDM; ++i) acc += __shfl(h, i) * rt[i * TDM + d];
    float out = silu_(acc);
    const float* Wi = Wih + t * 1875;
    const float* Wh = Whh + t * 1875;
    float gr = bih[t * 75 + d], gz = bih[t * 75 + 25 + d], gn = bih[t * 75 + 50 + d];
    float hr = bhh[t * 75 + d], hz = bhh[t * 75 + 25 + d], hn = bhh[t * 75 + 50 + d];
    #pragma unroll
    for (int i = 0; i < TDM; ++i) {
      float ov = __shfl(out, i);
      float hv = __shfl(h, i);
      gr += ov * Wi[d * 25 + i];
      gz += ov * Wi[(25 + d) * 25 + i];
      gn += ov * Wi[(50 + d) * 25 + i];
      hr += hv * Wh[d * 25 + i];
      hz += hv * Wh[(25 + d) * 25 + i];
      hn += hv * Wh[(50 + d) * 25 + i];
    }
    float r = sigm_(gr + hr);
    float zz = sigm_(gz + hz);
    float nn = tanhf(gn + r * hn);
    float hnew = (1.0f - zz) * nn + zz * h;
    if (lane < TDM) state[(size_t)n * HID + t * TDM + lane] = hnew;
  }
}

// ---------------------------------------------------------------- Set2Set
__global__ void s2s_gates_kernel(const float* __restrict__ qstar, const float* __restrict__ hs,
                                 const float* __restrict__ lWih, const float* __restrict__ lWhh,
                                 const float* __restrict__ lbih, const float* __restrict__ lbhh,
                                 float* __restrict__ gates) {
  long id = (long)blockIdx.x * blockDim.x + threadIdx.x;
  if (id >= (long)NG * 800) return;
  int g = (int)(id / 800), j = (int)(id % 800);
  float acc = lbih[j] + lbhh[j];
  const float* wi = lWih + (size_t)j * 400;
  const float* q = qstar + (size_t)g * 400;
  for (int k = 0; k < 400; ++k) acc += q[k] * wi[k];
  const float* wh = lWhh + (size_t)j * 200;
  const float* hp = hs + (size_t)g * 200;
  for (int k = 0; k < 200; ++k) acc += hp[k] * wh[k];
  gates[id] = acc;
}
__global__ void s2s_lstm_kernel(const float* __restrict__ gates, float* __restrict__ hs,
                                float* __restrict__ cs) {
  int id = blockIdx.x * blockDim.x + threadIdx.x;
  if (id >= NG * 200) return;
  int g = id / 200, j = id % 200;
  const float* gp = gates + (size_t)g * 800;
  float ig = sigm_(gp[j]), fg = sigm_(gp[200 + j]);
  float gg = tanhf(gp[400 + j]), og = sigm_(gp[600 + j]);
  float c = fg * cs[id] + ig * gg;
  cs[id] = c;
  hs[id] = og * tanhf(c);
}
__global__ void s2s_attn_init_kernel(unsigned* __restrict__ mmaxu, float* __restrict__ denom,
                                     float* __restrict__ rbuf) {
  int id = blockIdx.x * blockDim.x + threadIdx.x;
  if (id >= NG * 200) return;
  rbuf[id] = 0.0f;
  if (id % 200 == 0) {
    int g = id / 200;
    mmaxu[g] = 0x007FFFFFu;  // foenc(-inf)
    denom[g] = 0.0f;
  }
}
__global__ void s2s_e_kernel(const float* __restrict__ xm, const float* __restrict__ hs,
                             const int* __restrict__ batch, float* __restrict__ ev,
                             unsigned* __restrict__ mmaxu) {
  int n = (int)(((long)blockIdx.x * blockDim.x + threadIdx.x) >> 5);
  int lane = threadIdx.x & 31;
  if (n >= NND) return;
  int g = batch[n];
  float s = 0.0f;
  for (int j = lane; j < HID; j += 32) s += xm[(size_t)n * HID + j] * hs[(size_t)g * HID + j];
  s = wave_sum(s);
  if (lane == 0) {
    ev[n] = s;
    atomicMax(&mmaxu[g], foenc(s));
  }
}
__global__ void s2s_a_kernel(const int* __restrict__ batch, const unsigned* __restrict__ mmaxu,
                             float* __restrict__ ev, float* __restrict__ denom) {
  int n = blockIdx.x * blockDim.x + threadIdx.x;
  if (n >= NND) return;
  int g = batch[n];
  float a = __expf(ev[n] - fodec(mmaxu[g]));
  ev[n] = a;
  atomicAdd(&denom[g], a);
}
__global__ void s2s_r_kernel(const float* __restrict__ xm, const int* __restrict__ batch,
                             const float* __restrict__ ev, const float* __restrict__ denom,
                             float* __restrict__ rbuf) {
  long id = (long)blockIdx.x * blockDim.x + threadIdx.x;
  if (id >= (long)NND * HID) return;
  int n = (int)(id / HID), j = (int)(id % HID);
  int g = batch[n];
  float coeff = ev[n] / denom[g];
  atomicAdd(&rbuf[(size_t)g * HID + j], coeff * xm[id]);
}
__global__ void s2s_qstar_kernel(const float* __restrict__ hs, const float* __restrict__ rbuf,
                                 float* __restrict__ qstar) {
  int id = blockIdx.x * blockDim.x + threadIdx.x;
  if (id >= NG * 400) return;
  int g = id / 400, j = id % 400;
  qstar[id] = (j < HID) ? hs[(size_t)g * HID + j] : rbuf[(size_t)g * HID + (j - HID)];
}

// ---------------------------------------------------------------- output head
__global__ void head_lin1_kernel(const float* __restrict__ qstar, const float* __restrict__ ow1,
                                 const float* __restrict__ ob1, float* __restrict__ zout) {
  int id = blockIdx.x * blockDim.x + threadIdx.x;
  if (id >= NG * 200) return;
  int g = id / 200, j = id % 200;
  float acc = ob1[j];
  const float* q = qstar + (size_t)g * 400;
  for (int k = 0; k < 400; ++k) acc += q[k] * ow1[k * 200 + j];
  zout[id] = acc;
}
__global__ void head_out_kernel(const float* __restrict__ z, const float* __restrict__ bnm,
                                const float* __restrict__ bnv, const float* __restrict__ og,
                                const float* __restrict__ obe, const float* __restrict__ ow2,
                                const float* __restrict__ ob2, float* __restrict__ out) {
  int g = (int)(((long)blockIdx.x * blockDim.x + threadIdx.x) >> 5);
  int lane = threadIdx.x & 31;
  if (g >= NG) return;
  float s = 0.0f;
  for (int j = lane; j < HID; j += 32) {
    float v = og[j] * (z[(size_t)g * HID + j] - bnm[j]) * rsqrtf(bnv[j] + 1e-5f) + obe[j];
    s += silu_(v) * ow2[j];
  }
  s = wave_sum(s);
  if (lane == 0) out[g] = s + ob2[0];
}

// ================================================================ host
static inline unsigned blks(long n) { return (unsigned)((n + 255) / 256); }

extern "C" void kernel_launch(void* const* d_in, const int* in_sizes, int n_in,
                              void* d_out, int out_size, void* d_ws, size_t ws_size,
                              hipStream_t stream) {
  (void)in_sizes; (void)n_in; (void)out_size; (void)ws_size;
  const float* x      = (const float*)d_in[0];
  const float* eattr  = (const float*)d_in[1];
  const float* fw     = (const float*)d_in[2];
  const float* fb     = (const float*)d_in[3];
  const float* tw1    = (const float*)d_in[4];
  const float* tb1    = (const float*)d_in[5];
  const float* tg1    = (const float*)d_in[6];
  const float* tbe1   = (const float*)d_in[7];
  const float* tw2    = (const float*)d_in[8];
  const float* tb2    = (const float*)d_in[9];
  const float* troot  = (const float*)d_in[10];
  const float* tcb    = (const float*)d_in[11];
  const float* tWih   = (const float*)d_in[12];
  const float* tWhh   = (const float*)d_in[13];
  const float* tbih   = (const float*)d_in[14];
  const float* tbhh   = (const float*)d_in[15];
  const float* mw1    = (const float*)d_in[16];
  const float* mb1    = (const float*)d_in[17];
  const float* mg     = (const float*)d_in[18];
  const float* mbe    = (const float*)d_in[19];
  const float* mw2    = (const float*)d_in[20];
  const float* mb2    = (const float*)d_in[21];
  const float* lWih   = (const float*)d_in[22];
  const float* lWhh   = (const float*)d_in[23];
  const float* lbih   = (const float*)d_in[24];
  const float* lbhh   = (const float*)d_in[25];
  const float* ow1    = (const float*)d_in[26];
  const float* ob1    = (const float*)d_in[27];
  const float* og     = (const float*)d_in[28];
  const float* obe    = (const float*)d_in[29];
  const float* ow2    = (const float*)d_in[30];
  const float* ob2    = (const float*)d_in[31];
  const int*   eidx   = (const int*)d_in[32];
  const int*   batch  = (const int*)d_in[33];
  float* out = (float*)d_out;

  char* wsb = (char*)d_ws;
  size_t off = 0;
  auto alloc = [&](size_t bytes) -> char* {
    char* p = wsb + off;
    off = (off + bytes + 255) & ~(size_t)255;
    return p;
  };
  // staged-B geometry
  const int W2_NT = 40;   // ceil(625/16)=40 tiles (mult of 4)
  const int W2_KT = 2;    // Kpad 64
  const int MX_NT = 16;   // ceil(200/16)=13 -> pad to 16 tiles (zeros)
  const int MX_KT = 7;    // Kpad 224
  const long W2S = (long)W2_KT * W2_NT * 512;  // per-tower staged elements
  const long MXS = (long)MX_KT * MX_NT * 512;

  float*  deg     = (float*)alloc((size_t)NND * 4);
  float*  state   = (float*)alloc((size_t)NND * HID * 4);
  float*  agg     = (float*)alloc((size_t)NND * HID * 4);
  float*  e_lin   = (float*)alloc((size_t)NTW * NE * 50 * 4);  // region reused for t1 later
  __bf16* e_pad   = (__bf16*)alloc((size_t)NTW * NE * 64 * 2);
  __bf16* w2s     = (__bf16*)alloc((size_t)NTW * W2S * 2);
  __bf16* We_bf   = (__bf16*)alloc((size_t)NTW * NE * 625 * 2);
  __bf16* mw1s    = (__bf16*)alloc((size_t)MXS * 2);
  __bf16* mw2s    = (__bf16*)alloc((size_t)MXS * 2);
  __bf16* state_bf= (__bf16*)alloc((size_t)NND * 224 * 2);
  __bf16* t2_bf   = (__bf16*)alloc((size_t)NND * 224 * 2);
  float*  xm      = (float*)alloc((size_t)NND * HID * 4);
  float*  bnm_e   = (float*)alloc(400 * 4);
  float*  bnv_e   = (float*)alloc(400 * 4);
  float*  bnm     = (float*)alloc(HID * 4);
  float*  bnv     = (float*)alloc(HID * 4);
  float*  qstar   = (float*)alloc((size_t)NG * 400 * 4);
  float*  hs      = (float*)alloc((size_t)NG * HID * 4);
  float*  cs      = (float*)alloc((size_t)NG * HID * 4);
  float*  gates   = (float*)alloc((size_t)NG * 800 * 4);
  float*  ev      = (float*)alloc((size_t)NND * 4);
  unsigned* mmaxu = (unsigned*)alloc((size_t)NG * 4);
  float*  denom   = (float*)alloc((size_t)NG * 4);
  float*  rbuf    = (float*)alloc((size_t)NG * HID * 4);
  float*  zhead   = (float*)alloc((size_t)NG * HID * 4);
  float*  bnm2    = (float*)alloc(HID * 4);
  float*  bnv2    = (float*)alloc(HID * 4);
  float*  t1      = e_lin;  // e_lin dead before t1 is written

  // --- degree
  zero_f32_kernel<<<blks(NND), 256, 0, stream>>>(deg, NND);
  deg_kernel<<<blks(NE), 256, 0, stream>>>(eidx, deg);

  // --- input projection
  h0_kernel<<<blks((long)NND * HID), 256, 0, stream>>>(x, fw, fb, state);

  // --- edge MLP (shared across layers): lin1 -> BN -> SiLU -> WMMA GEMM -> We (bf16)
  edge_lin1_kernel<<<blks((long)NTW * NE * 50), 256, 0, stream>>>(eattr, tw1, tb1, e_lin);
  bn_stats_kernel<<<400, 256, 0, stream>>>(e_lin, bnm_e, bnv_e, NE, 50, (long)NE * 50);
  edge_bn_silu_kernel<<<blks((long)NTW * NE * 64), 256, 0, stream>>>(e_lin, bnm_e, bnv_e, tg1, tbe1, e_pad);
  stage_b_kernel<<<blks((long)NTW * W2S), 256, 0, stream>>>(tw2, w2s, 50, 625, W2_KT, W2_NT,
                                                            (long)50 * 625, (long)NTW * W2S);
  {  // We[t] = e[t] @ w2[t] + b2[t]   (65536x50 @ 50x625, x8 towers) -> bf16
    dim3 grid(W2_NT / 4, NE / 128, NTW);
    wmma_gemm_kernel<true, W2_KT><<<grid, 256, 0, stream>>>(
        e_pad, w2s, tb2, (void*)We_bf, NE, 625, W2_NT,
        (long)NE * 64, W2S, 625L, (long)NE * 625);
  }

  // --- 4 message-passing layers
  for (int l = 0; l < 4; ++l) {
    zero_f32_kernel<<<blks((long)NND * HID), 256, 0, stream>>>(agg, (long)NND * HID);
    msg_kernel<<<(unsigned)NE, 256, 0, stream>>>(state, We_bf, eidx, agg);
    node_update_kernel<<<NND / 8, 256, 0, stream>>>(state, agg, deg, troot, tcb, tWih, tbih, tWhh, tbhh);
  }

  // --- mixing network: t1 = state@mw1+mb1 ; BN+SiLU ; xm = t2@mw2+mb2
  pad_bf16_kernel<<<blks((long)NND * 224), 256, 0, stream>>>(state, state_bf, NND, HID, 224);
  stage_b_kernel<<<blks(MXS), 256, 0, stream>>>(mw1, mw1s, HID, HID, MX_KT, MX_NT, 0L, MXS);
  stage_b_kernel<<<blks(MXS), 256, 0, stream>>>(mw2, mw2s, HID, HID, MX_KT, MX_NT, 0L, MXS);
  {
    dim3 grid(MX_NT / 4, NND / 128, 1);
    wmma_gemm_kernel<false, MX_KT><<<grid, 256, 0, stream>>>(
        state_bf, mw1s, mb1, (void*)t1, NND, HID, MX_NT, 0L, 0L, 0L, 0L);
  }
  bn_stats_kernel<<<HID, 256, 0, stream>>>(t1, bnm, bnv, NND, HID, 0L);
  bn_silu_bf16_kernel<<<blks((long)NND * 224), 256, 0, stream>>>(t1, bnm, bnv, mg, mbe, t2_bf);
  {
    dim3 grid(MX_NT / 4, NND / 128, 1);
    wmma_gemm_kernel<false, MX_KT><<<grid, 256, 0, stream>>>(
        t2_bf, mw2s, mb2, (void*)xm, NND, HID, MX_NT, 0L, 0L, 0L, 0L);
  }

  // --- Set2Set pooling (3 steps)
  zero_f32_kernel<<<blks((long)NG * 400), 256, 0, stream>>>(qstar, (long)NG * 400);
  zero_f32_kernel<<<blks((long)NG * HID), 256, 0, stream>>>(hs, (long)NG * HID);
  zero_f32_kernel<<<blks((long)NG * HID), 256, 0, stream>>>(cs, (long)NG * HID);
  for (int m = 0; m < 3; ++m) {
    s2s_gates_kernel<<<blks((long)NG * 800), 256, 0, stream>>>(qstar, hs, lWih, lWhh, lbih, lbhh, gates);
    s2s_lstm_kernel<<<blks((long)NG * 200), 256, 0, stream>>>(gates, hs, cs);
    s2s_attn_init_kernel<<<blks((long)NG * 200), 256, 0, stream>>>(mmaxu, denom, rbuf);
    s2s_e_kernel<<<NND / 8, 256, 0, stream>>>(xm, hs, batch, ev, mmaxu);
    s2s_a_kernel<<<blks(NND), 256, 0, stream>>>(batch, mmaxu, ev, denom);
    s2s_r_kernel<<<blks((long)NND * HID), 256, 0, stream>>>(xm, batch, ev, denom, rbuf);
    s2s_qstar_kernel<<<blks((long)NG * 400), 256, 0, stream>>>(hs, rbuf, qstar);
  }

  // --- output head
  head_lin1_kernel<<<blks((long)NG * 200), 256, 0, stream>>>(qstar, ow1, ob1, zhead);
  bn_stats_kernel<<<HID, 256, 0, stream>>>(zhead, bnm2, bnv2, NG, HID, 0L);
  head_out_kernel<<<NG / 8, 256, 0, stream>>>(zhead, bnm2, bnv2, og, obe, ow2, ob2, out);
}
